// TimeLSTM_88768384074097
// MI455X (gfx1250) — matrix operand order repaired and checked by reading, not compiled, not run on hardware
//
#include <hip/hip_runtime.h>
#include <hip/hip_bf16.h>
#include <math.h>

#define T_STEPS 32768
#define D_INP   256
#define HDIM    256
#define NG      1024   // 4*H concatenated gates [i|f|o|c]

typedef __attribute__((ext_vector_type(16))) __bf16 v16bf;
typedef __attribute__((ext_vector_type(8)))  float  v8f;

__device__ __forceinline__ unsigned short f2bf(float f) {
    unsigned u = __float_as_uint(f);
    unsigned r = (u + 0x7FFFu + ((u >> 16) & 1u)) >> 16;  // RNE
    return (unsigned short)r;
}
__device__ __forceinline__ float bf2f(unsigned short h) {
    return __uint_as_float(((unsigned)h) << 16);
}

// ---------------- init: zero h/c ping-pong buffers + sync counter -----------
__global__ void tl_init_kernel(float* hbuf, float* cbuf, unsigned* cnt) {
    int i = threadIdx.x;
    if (i < 512) { hbuf[i] = 0.0f; cbuf[i] = 0.0f; }
    if (i == 0) *cnt = 0u;
}

// ---------------- packing kernels (fp32 -> bf16 layouts) --------------------
__global__ void tl_pack_inputs(const float* __restrict__ in,
                               unsigned short* __restrict__ Abf, int n) {
    int i = blockIdx.x * blockDim.x + threadIdx.x;
    if (i < n) Abf[i] = f2bf(in[i]);
}

// Bt[n][k] = Wcat[k][n] transposed, bf16.  n in [0,1024): g=n>>8, j=n&255.
__global__ void tl_pack_wcat(const float* __restrict__ Wi, const float* __restrict__ Wf,
                             const float* __restrict__ Wo, const float* __restrict__ Wc,
                             unsigned short* __restrict__ Bt) {
    int i = blockIdx.x * blockDim.x + threadIdx.x;   // over 1024*256
    if (i >= NG * D_INP) return;
    int n = i >> 8;
    int k = i & 255;
    int g = n >> 8, j = n & 255;
    const float* W = (g == 0) ? Wi : (g == 1) ? Wf : (g == 2) ? Wo : Wc;
    Bt[n * D_INP + k] = f2bf(W[k * HDIM + j]);
}

__global__ void tl_pack_bias(const float* __restrict__ bi, const float* __restrict__ bff,
                             const float* __restrict__ bo, const float* __restrict__ bc,
                             float* __restrict__ bcat) {
    int i = blockIdx.x * blockDim.x + threadIdx.x;
    if (i >= NG) return;
    int g = i >> 8, j = i & 255;
    const float* b = (g == 0) ? bi : (g == 1) ? bff : (g == 2) ? bo : bc;
    bcat[i] = b[j];
}

// Recurrent weight columns, bf16-pair packed, colId = j*5 + which
// which: 0..3 -> U_i,U_f,U_o,U_c ; 4 -> W_d   (column j of each)
__global__ void tl_pack_ucat(const float* __restrict__ Ui, const float* __restrict__ Uf,
                             const float* __restrict__ Uo, const float* __restrict__ Uc,
                             const float* __restrict__ Wd, unsigned* __restrict__ ucat) {
    int i = blockIdx.x * blockDim.x + threadIdx.x;   // over 1280*128
    if (i >= 1280 * 128) return;
    int colId = i >> 7, kk = i & 127;
    int j = colId / 5, which = colId % 5;
    const float* M = (which == 0) ? Ui : (which == 1) ? Uf :
                     (which == 2) ? Uo : (which == 3) ? Uc : Wd;
    unsigned lo = f2bf(M[(2 * kk) * HDIM + j]);
    unsigned hi = f2bf(M[(2 * kk + 1) * HDIM + j]);
    ucat[colId * 128 + kk] = lo | (hi << 16);
}

// ---------------- WMMA GEMM: gates[T,1024] = Abf @ Wcat + bcat --------------
// One 16x64 output tile per wave (4 accumulators): the A fragment is reused
// across 4 WMMAs per K-step (4 independent accumulation chains interleave),
// K-loop of 8 -> 32 x wmma_f32_16x16x32_bf16 per wave.
__global__ __launch_bounds__(256)
void tl_gemm_kernel(const unsigned short* __restrict__ Abf,   // [T,256] bf16 row-major
                    const unsigned short* __restrict__ Bt,    // [1024,256] bf16 (N-major)
                    const float* __restrict__ bcat,           // [1024]
                    float* __restrict__ gates) {              // [T,1024] fp32
    const int lane  = threadIdx.x & 31;
    const int wave  = threadIdx.x >> 5;
    const int m0    = blockIdx.x * 16;
    const int n0    = (blockIdx.y * 8 + wave) * 64;           // 16 n-groups of 64
    const int rowA  = m0 + (lane & 15);
    const int koffA = (lane >> 4) * 8;    // ISA A layout: lanes 16-31 hold K+8 runs
    const int colB0 = n0 + (lane & 15);
    const int koffB = (lane >> 4) * 8;    // B layout matches A: lanes 16-31 hold K+8 runs
    const unsigned short* aRow = Abf + (size_t)rowA * D_INP;
    const unsigned short* bCol0 = Bt + (size_t)colB0 * D_INP;

    v8f acc[4] = {{}, {}, {}, {}};
#pragma unroll
    for (int kk = 0; kk < 8; ++kk) {
        const int k0 = kk * 32;
        union { uint4 q[2]; v16bf v; } a;
        a.q[0] = *(const uint4*)(aRow + k0 + koffA);        // K = k0+koffA .. +7
        a.q[1] = *(const uint4*)(aRow + k0 + 16 + koffA);   // K = k0+16+koffA .. +7
#pragma unroll
        for (int j = 0; j < 4; ++j) {
            const unsigned short* bCol = bCol0 + (size_t)(16 * j) * D_INP;
            union { uint4 q[2]; v16bf v; } b;
            b.q[0] = *(const uint4*)(bCol + k0 + koffB);    // K contiguous per column
            b.q[1] = *(const uint4*)(bCol + k0 + 16 + koffB);
            acc[j] = __builtin_amdgcn_wmma_f32_16x16x32_bf16(
                         /*neg_a=*/false, a.v, /*neg_b=*/false, b.v,
                         /*c_mod=*/(short)0, acc[j],
                         /*reuse_a=*/false, /*reuse_b=*/false);
        }
    }
    const int rbase = m0 + (lane >> 4) * 8;
#pragma unroll
    for (int j = 0; j < 4; ++j) {
        const int colB = colB0 + 16 * j;
        const float bias = bcat[colB];
#pragma unroll
        for (int v = 0; v < 8; ++v) {
            gates[(size_t)(rbase + v) * NG + colB] = acc[j][v] + bias;
        }
    }
}

// ---------------- serial scan: 16 WGs x 160 threads, LDS-resident weights ---
// WG wg owns hidden indices j in [wg*16, wg*16+16): for each j the 4 U columns
// and the W_d column (80 dot products of length 256).  h/c exchanged through
// L2 each step, guarded by a monotonic device-scope counter barrier.
__global__ __launch_bounds__(160)
void tl_scan_kernel(const float* __restrict__ gates,      // [T,1024]
                    const unsigned* __restrict__ ucat,    // [1280,128] bf16 pairs
                    const int* __restrict__ created_at,   // [T]
                    const float* __restrict__ b_d,        // [256]
                    float* __restrict__ hbuf,             // [2][256]
                    float* __restrict__ cbuf,             // [2][256]
                    unsigned* __restrict__ cnt,
                    float* __restrict__ out) {            // [T,256]
    __shared__ unsigned wlds[80 * 129];   // padded stride 129 -> no bank collide
    __shared__ float h_lds[256];
    __shared__ float c_lds[256];
    __shared__ float red[160];
    __shared__ float val[80];

    const int tid = threadIdx.x;
    const int wg  = blockIdx.x;           // 0..15

    // Stage this WG's 80 weight columns into LDS (once).
    for (int i = tid; i < 80 * 128; i += 160) {
        int col = i >> 7, kk = i & 127;
        wlds[col * 129 + kk] = ucat[(size_t)(wg * 80 + col) * 128 + kk];
    }
    // Per-thread dot assignment: d in [0,64) gate dots (g=d>>4, l=d&15, vs h);
    // d in [64,80) W_d dots (vs c).  Threads 80-159 do the second K-half.
    const int d    = (tid < 80) ? tid : tid - 80;
    const int half = (tid < 80) ? 0 : 1;
    const int kb   = half * 64;           // packed-pair base index
    const int col  = (d < 64) ? ((d & 15) * 5 + (d >> 4)) : ((d - 64) * 5 + 4);
    const unsigned* wcol = &wlds[col * 129 + kb];
    __syncthreads();

    for (int t = 0; t < T_STEPS; ++t) {
        if (t > 0) {
            if (tid == 0) {
                while (__hip_atomic_load(cnt, __ATOMIC_ACQUIRE,
                                         __HIP_MEMORY_SCOPE_AGENT) < (unsigned)(16 * t))
                    __builtin_amdgcn_s_sleep(1);
            }
            __syncthreads();
        }
        {   // pull full h_prev / c_prev for this step from the ping-pong buffers
            const float* hsrc = hbuf + (t & 1) * 256;
            const float* csrc = cbuf + (t & 1) * 256;
            for (int i = tid; i < 256; i += 160) { h_lds[i] = hsrc[i]; c_lds[i] = csrc[i]; }
        }
        if (t + 1 < T_STEPS && tid < 4) {   // warm next step's gate row (global_prefetch_b8)
            __builtin_prefetch(&gates[(size_t)(t + 1) * NG + tid * 256 + wg * 16], 0, 1);
        }
        __syncthreads();

        const float* vec = (d < 64) ? h_lds : c_lds;
        float acc = 0.0f;
#pragma unroll 8
        for (int kk = 0; kk < 64; ++kk) {
            unsigned u = wcol[kk];
            int k = (kb + kk) * 2;
            acc += vec[k]     * bf2f((unsigned short)(u & 0xFFFFu));
            acc += vec[k + 1] * bf2f((unsigned short)(u >> 16));
        }
        red[tid] = acc;
        __syncthreads();

        if (tid < 80) {
            float tot = red[tid] + red[tid + 80];
            if (d < 64) {
                int g = d >> 4, l = d & 15;
                int j = wg * 16 + l;
                val[d] = tot + gates[(size_t)t * NG + g * 256 + j];  // x-proj carries bias
            } else {
                int j = wg * 16 + (d - 64);
                val[d] = tanhf(tot + b_d[j]);                        // s = tanh(c@W_d + b_d)
            }
        }
        __syncthreads();

        if (tid < 16) {
            const int l = tid, j = wg * 16 + l;
            float dtv   = (t == 0) ? 0.0f : (float)(created_at[t] - created_at[t - 1]);
            float decay = 1.0f / logf(2.718281828459045f + dtv);
            float ig = 1.0f / (1.0f + expf(-val[l]));
            float fg = 1.0f / (1.0f + expf(-val[16 + l]));
            float og = 1.0f / (1.0f + expf(-val[32 + l]));
            float gg = tanhf(val[48 + l]);
            float s  = val[64 + l];
            float cp = c_lds[j];
            float adj = (cp - s) + s * decay;
            float cn  = fg * adj + ig * gg;
            float hn  = og * tanhf(cp);                // uses PREV memory (faithful)
            out[(size_t)t * HDIM + j]     = hn;
            hbuf[((t + 1) & 1) * 256 + j] = hn;
            cbuf[((t + 1) & 1) * 256 + j] = cn;
        }
        __syncthreads();
        if (tid == 0) {
            __threadfence();
            __hip_atomic_fetch_add(cnt, 1u, __ATOMIC_RELEASE, __HIP_MEMORY_SCOPE_AGENT);
        }
    }
}

// ---------------------------------------------------------------------------
extern "C" void kernel_launch(void* const* d_in, const int* in_sizes, int n_in,
                              void* d_out, int out_size, void* d_ws, size_t ws_size,
                              hipStream_t stream) {
    const float* inputs     = (const float*)d_in[0];
    const int*   created_at = (const int*)  d_in[1];
    const float* W_d = (const float*)d_in[2];
    // d_in[3] = U_d (unused by reference)
    const float* b_d = (const float*)d_in[4];
    const float* W_f = (const float*)d_in[5];
    const float* U_f = (const float*)d_in[6];
    const float* b_f = (const float*)d_in[7];
    const float* W_i = (const float*)d_in[8];
    const float* U_i = (const float*)d_in[9];
    const float* b_i = (const float*)d_in[10];
    const float* W_o = (const float*)d_in[11];
    const float* U_o = (const float*)d_in[12];
    const float* b_o = (const float*)d_in[13];
    const float* W_c = (const float*)d_in[14];
    const float* U_c = (const float*)d_in[15];
    const float* b_c = (const float*)d_in[16];
    float* out = (float*)d_out;

    char* ws = (char*)d_ws;
    size_t off = 0;
    float*          gates = (float*)(ws + off);          off += (size_t)T_STEPS * NG * 4;     // 128 MB
    unsigned short* Abf   = (unsigned short*)(ws + off); off += (size_t)T_STEPS * D_INP * 2;  // 16 MB
    unsigned short* Bt    = (unsigned short*)(ws + off); off += (size_t)NG * D_INP * 2;       // 512 KB
    float*          bcat  = (float*)(ws + off);          off += NG * 4;
    unsigned*       ucat  = (unsigned*)(ws + off);       off += 1280 * 128 * 4;               // 640 KB
    float*          hbuf  = (float*)(ws + off);          off += 2 * 256 * 4;
    float*          cbuf  = (float*)(ws + off);          off += 2 * 256 * 4;
    unsigned*       cnt   = (unsigned*)(ws + off);       off += 128;

    tl_init_kernel<<<1, 512, 0, stream>>>(hbuf, cbuf, cnt);

    const int nA = T_STEPS * D_INP;
    tl_pack_inputs<<<(nA + 255) / 256, 256, 0, stream>>>(inputs, Abf, nA);
    tl_pack_wcat  <<<(NG * D_INP + 255) / 256, 256, 0, stream>>>(W_i, W_f, W_o, W_c, Bt);
    tl_pack_bias  <<<(NG + 255) / 256, 256, 0, stream>>>(b_i, b_f, b_o, b_c, bcat);
    tl_pack_ucat  <<<(1280 * 128 + 255) / 256, 256, 0, stream>>>(U_i, U_f, U_o, U_c, W_d, ucat);

    // GEMM: 2048 m-tiles x (2 n-groups x 8 waves) -> each wave a 16x64 tile
    tl_gemm_kernel<<<dim3(2048, 2, 1), 256, 0, stream>>>(Abf, Bt, bcat, gates);

    // Serial scan: 16 cooperating workgroups
    tl_scan_kernel<<<16, 160, 0, stream>>>(gates, ucat, created_at, b_d,
                                           hbuf, cbuf, cnt, out);
}
